// HighwayBiLSTM_19404662243373
// MI455X (gfx1250) — compile-verified, matrix operand order checked by reading, not run on hardware
//
#include <hip/hip_runtime.h>

typedef __attribute__((ext_vector_type(16))) _Float16 v16h;
typedef __attribute__((ext_vector_type(8)))  float    v8f;

#define HDIM  512
#define TSEQ  512
#define BATCH 32
#define GDIM  3072   // 6H
#define NREC  2560   // 5H
#define KDIM  512
#define MROWS (TSEQ*BATCH)   // 16384

// ------------- CDNA5 async global->LDS copy (guarded, with fallback) -------------

#if __has_builtin(__builtin_amdgcn_global_load_async_to_lds_b128)
#define HAVE_ASYNC_LDS 1
#else
#define HAVE_ASYNC_LDS 0
#endif

typedef int v4i __attribute__((vector_size(16)));
typedef __attribute__((address_space(1))) v4i gv4i;   // global
typedef __attribute__((address_space(3))) v4i sv4i;   // LDS

// copy 16 bytes global -> LDS
__device__ __forceinline__ void cp16_g2l(const _Float16* g, _Float16* l) {
#if HAVE_ASYNC_LDS
  __builtin_amdgcn_global_load_async_to_lds_b128((gv4i*)g, (sv4i*)l, 0, 0);
#else
  *(uint4*)l = *(const uint4*)g;
#endif
}

__device__ __forceinline__ void cp_wait() {
#if HAVE_ASYNC_LDS
#if __has_builtin(__builtin_amdgcn_s_wait_asynccnt)
  __builtin_amdgcn_s_wait_asynccnt(0);
#else
  asm volatile("s_wait_asynccnt 0x0" ::: "memory");
#endif
#endif
}

// ---------------- utility kernels ----------------

__global__ void cvt_f32_to_f16(const float* __restrict__ s, _Float16* __restrict__ d, int n) {
  int i = blockIdx.x * blockDim.x + threadIdx.x;
  int stride = gridDim.x * blockDim.x;
  for (; i < n; i += stride) d[i] = (_Float16)s[i];
}

__global__ void zero_f16(_Float16* d, int n) {
  int i = blockIdx.x * blockDim.x + threadIdx.x;
  int stride = gridDim.x * blockDim.x;
  for (; i < n; i += stride) d[i] = (_Float16)0.0f;
}

__global__ void zero_i32(int* d, int n) {
  int i = blockIdx.x * blockDim.x + threadIdx.x;
  if (i < n) d[i] = 0;
}

// ---------------- xproj GEMM: out[M,N] = A[M,K] * W[N,K]^T + bias ----------------
// 64x64 tile per 256-thread block, 8 waves, each wave -> 16x32 (two 16x16 WMMA tiles).

__launch_bounds__(256)
__global__ void gemm_xproj(const _Float16* __restrict__ A,   // [M,K] f16 row-major
                           const _Float16* __restrict__ W,   // [N,K] f16 row-major
                           const float*    __restrict__ bias,// [N]
                           float*          __restrict__ out, // [M,N] f32
                           int M, int N, int K) {
  __shared__ __align__(32) _Float16 As[64 * 32];
  __shared__ __align__(32) _Float16 Bs[64 * 32];

  const int tid  = threadIdx.x;
  const int lane = tid & 31;
  const int w    = tid >> 5;            // wave 0..7
  const int m0   = blockIdx.x * 64;
  const int n0   = blockIdx.y * 64;
  const int wm   = (w & 3) * 16;        // wave M offset inside tile
  const int wn   = (w >> 2) * 32;       // wave N offset inside tile

  // staging: each thread moves 8 f16 (16B) of A and of B per k-tile
  const int e  = tid * 8;
  const int lr = e >> 5;                // 0..63 (tile row)
  const int lk = e & 31;                // 0..31 (tile k)

  v8f acc0 = {};
  v8f acc1 = {};

  for (int k0 = 0; k0 < K; k0 += 32) {
    __syncthreads();
    cp16_g2l(&A[(size_t)(m0 + lr) * K + k0 + lk], &As[lr * 32 + lk]);
    cp16_g2l(&W[(size_t)(n0 + lr) * K + k0 + lk], &Bs[lr * 32 + lk]);
    if (k0 + 32 < K) {
      __builtin_prefetch(&A[(size_t)(m0 + lr) * K + k0 + 32 + lk], 0, 1);
      __builtin_prefetch(&W[(size_t)(n0 + lr) * K + k0 + 32 + lk], 0, 1);
    }
    cp_wait();
    __syncthreads();

    const int krow = (lane >> 4) * 16;
    const int ml   = lane & 15;
    v16h a  = *(const v16h*)&As[(wm + ml) * 32 + krow];
    v16h b0 = *(const v16h*)&Bs[(wn + ml) * 32 + krow];
    v16h b1 = *(const v16h*)&Bs[(wn + 16 + ml) * 32 + krow];
    acc0 = __builtin_amdgcn_wmma_f32_16x16x32_f16(false, a, false, b0, (short)0, acc0, false, false);
    acc1 = __builtin_amdgcn_wmma_f32_16x16x32_f16(false, a, false, b1, (short)0, acc1, false, false);
  }

  // C/D layout: lanes 0-15 -> M=r, N=lane; lanes 16-31 -> M=8+r, N=lane-16
  const int cn   = lane & 15;
  const int rofs = (lane >> 4) * 8;
#pragma unroll
  for (int r = 0; r < 8; ++r) {
    int gm  = m0 + wm + rofs + r;
    int gn0 = n0 + wn + cn;
    int gn1 = gn0 + 16;
    out[(size_t)gm * N + gn0] = acc0[r] + bias[gn0];
    out[(size_t)gm * N + gn1] = acc1[r] + bias[gn1];
  }
}

// ---------------- grid-wide barrier for the persistent scan kernel ----------------

__device__ __forceinline__ void grid_barrier(int* cnt, int* gen, int nblocks) {
  __syncthreads();
  if (threadIdx.x == 0) {
    __threadfence();
    int g = __atomic_load_n(gen, __ATOMIC_ACQUIRE);
    if (atomicAdd(cnt, 1) == nblocks - 1) {
      __atomic_store_n(cnt, 0, __ATOMIC_RELAXED);
      __atomic_fetch_add(gen, 1, __ATOMIC_RELEASE);
    } else {
      while (__atomic_load_n(gen, __ATOMIC_ACQUIRE) == g)
        __builtin_amdgcn_s_sleep(2);
    }
    __threadfence();
  }
  __syncthreads();
}

__device__ __forceinline__ float sigmoidf_(float x) {
  return 1.0f / (1.0f + __expf(-x));
}

// ---------------- persistent recurrent scan (one directional layer) ----------------
// 32 blocks x 320 threads. Block b owns h-columns [16b, 16b+16).
// 10 waves: wave w -> (mi = w&1) batch half-tile, (gi = w>>1) gate in {i,f,o,t,j}.
// Dynamic LDS: whh slice (160 rows x 512 f16 = 160KB) pinned for the whole scan,
// plus a 32KB staging buffer for the h state refreshed each timestep.

#define WHH_ROWS   160                  // 5 gates x 16 columns
#define DYN_HALF   ((WHH_ROWS + BATCH) * HDIM)  // f16 elements of dynamic LDS
#define DYN_BYTES  (DYN_HALF * 2)

__launch_bounds__(320)
__global__ void recurrent_scan(const float*    __restrict__ xproj,  // [T*B, 6H] f32
                               const float*    __restrict__ masks,  // [T*B]
                               const _Float16* __restrict__ whh,    // [5H, H] f16
                               _Float16*       __restrict__ hq,     // [B, H] f16 state (pre-zeroed)
                               _Float16*       __restrict__ seqOut, // [T*B, H] f16
                               float*          __restrict__ outF32, // [T*B, H] f32 or null
                               float*          __restrict__ hN,     // [B, H] f32
                               float*          __restrict__ cN,     // [B, H] f32
                               int reverse,
                               int* syncCnt, int* syncGen) {
  extern __shared__ __align__(32) _Float16 dynS[];
  _Float16* whhS = dynS;                   // [160][512]
  _Float16* hA   = dynS + WHH_ROWS * HDIM; // [32][512]

  __shared__ float gS[5][32][16];   // activated gate tiles
  __shared__ float cS[32][16];      // cell state (this block's columns)
  __shared__ float hS[32][16];      // hidden state f32 (for h_n output)

  const int tid     = threadIdx.x;
  const int lane    = tid & 31;
  const int w       = tid >> 5;          // 0..9
  const int colbase = blockIdx.x * 16;   // h-column chunk
  const int mi      = w & 1;             // batch half (rows 0-15 / 16-31)
  const int gi      = w >> 1;            // gate index 0..4

  for (int i = tid; i < 512; i += 320) {
    cS[i >> 4][i & 15] = 0.0f;
    hS[i >> 4][i & 15] = 0.0f;
  }

  // pin this block's whh slice in LDS for the whole scan:
  // LDS row r = g*16 + j  <->  global whh row g*H + colbase + j
  for (int c = tid; c < WHH_ROWS * (HDIM / 8); c += 320) {
    int r  = c >> 6;          // 0..159
    int cc = (c & 63) * 8;    // 0..504 step 8
    int g  = r >> 4;
    int j  = r & 15;
    cp16_g2l(&whh[((size_t)g * HDIM + colbase + j) * HDIM + cc], &whhS[(size_t)r * HDIM + cc]);
  }
  cp_wait();
  __syncthreads();

  const int ml   = lane & 15;
  const int arow = mi * 16 + ml;
  const int brow = gi * 16 + ml;        // row in whhS
  const int krow = (lane >> 4) * 16;
  const int cn   = lane & 15;
  const int rofs = (lane >> 4) * 8;

  for (int s = 0; s < TSEQ; ++s) {
    const int t = reverse ? (TSEQ - 1 - s) : s;

    // stage current h (32x512 f16 = 32KB) from global into LDS
    for (int c = tid; c < BATCH * (HDIM / 8); c += 320)
      cp16_g2l(&hq[(size_t)c * 8], &hA[(size_t)c * 8]);
    cp_wait();
    __syncthreads();

    // preact tile = h @ whh^T   (16x16 per wave, K=512 in 16 WMMA steps, LDS-fed)
    v8f acc = {};
#pragma unroll 4
    for (int k0 = 0; k0 < HDIM; k0 += 32) {
      v16h a = *(const v16h*)&hA[(size_t)arow * HDIM + k0 + krow];
      v16h b = *(const v16h*)&whhS[(size_t)brow * HDIM + k0 + krow];
      acc = __builtin_amdgcn_wmma_f32_16x16x32_f16(false, a, false, b, (short)0, acc, false, false);
    }

    // add xproj, apply per-gate activation, publish to LDS
#pragma unroll
    for (int r = 0; r < 8; ++r) {
      int   m    = mi * 16 + rofs + r;                 // batch row 0..31
      int   gcol = gi * HDIM + colbase + cn;
      float p    = acc[r] + xproj[(size_t)(t * BATCH + m) * GDIM + gcol];
      float v;
      if (gi == 4)      v = tanhf(p);                   // j
      else if (gi == 1) v = sigmoidf_(p + 1.0f);        // f (+1 bias)
      else              v = sigmoidf_(p);               // i, o, t
      gS[gi][m][cn] = v;
    }
    __syncthreads();

    // elementwise cell/hidden update for this block's 32x16 chunk
    for (int i = tid; i < 512; i += 320) {
      int   m    = i >> 4, cc = i & 15;
      float iv   = gS[0][m][cc];
      float fv   = gS[1][m][cc];
      float ov   = gS[2][m][cc];
      float tv   = gS[3][m][cc];
      float jv   = gS[4][m][cc];
      float kv   = xproj[(size_t)(t * BATCH + m) * GDIM + 5 * HDIM + colbase + cc];
      float mval = masks[t * BATCH + m];
      float cold = cS[m][cc];
      float cnew = fv * cold + iv * jv;
      cnew = mval * cnew + (1.0f - mval) * cold;
      float hold = hS[m][cc];
      float hnew = tv * ov * tanhf(cnew) + (1.0f - tv) * kv;
      hnew = mval * hnew + (1.0f - mval) * hold;
      cS[m][cc] = cnew;
      hS[m][cc] = hnew;
      _Float16 hh = (_Float16)hnew;
      hq[(size_t)m * HDIM + colbase + cc] = hh;
      seqOut[(size_t)(t * BATCH + m) * HDIM + colbase + cc] = hh;
      if (outF32) outF32[(size_t)(t * BATCH + m) * HDIM + colbase + cc] = hnew;
    }

    // publish new h to all 32 blocks before the next step's staging reads
    grid_barrier(syncCnt, syncGen, gridDim.x);
  }

  // final h_T / c_T for this layer
  for (int i = tid; i < 512; i += 320) {
    int m = i >> 4, cc = i & 15;
    hN[(size_t)m * HDIM + colbase + cc] = hS[m][cc];
    cN[(size_t)m * HDIM + colbase + cc] = cS[m][cc];
  }
}

// ---------------- launch ----------------

extern "C" void kernel_launch(void* const* d_in, const int* in_sizes, int n_in,
                              void* d_out, int out_size, void* d_ws, size_t ws_size,
                              hipStream_t stream) {
  (void)in_sizes; (void)n_in; (void)out_size; (void)ws_size;

  const float* x     = (const float*)d_in[0];
  const float* masks = (const float*)d_in[1];
  const float* wih[4] = {(const float*)d_in[2],  (const float*)d_in[5],
                         (const float*)d_in[8],  (const float*)d_in[11]};
  const float* bih[4] = {(const float*)d_in[3],  (const float*)d_in[6],
                         (const float*)d_in[9],  (const float*)d_in[12]};
  const float* whh[4] = {(const float*)d_in[4],  (const float*)d_in[7],
                         (const float*)d_in[10], (const float*)d_in[13]};

  char*  ws  = (char*)d_ws;
  size_t off = 0;
  _Float16* wihh[4];
  for (int l = 0; l < 4; ++l) { wihh[l] = (_Float16*)(ws + off); off += (size_t)GDIM * KDIM * 2; }
  _Float16* whhh[4];
  for (int l = 0; l < 4; ++l) { whhh[l] = (_Float16*)(ws + off); off += (size_t)NREC * KDIM * 2; }
  _Float16* seqA = (_Float16*)(ws + off); off += (size_t)MROWS * HDIM * 2;
  _Float16* seqB = (_Float16*)(ws + off); off += (size_t)MROWS * HDIM * 2;
  float*   xproj = (float*)(ws + off);    off += (size_t)MROWS * GDIM * 4;
  _Float16* hq   = (_Float16*)(ws + off); off += (size_t)BATCH * HDIM * 2;
  int*     sync  = (int*)(ws + off);      off += 256;

  // allow >64KB dynamic LDS for the persistent scan kernel (no-op if already allowed)
  static int lds_attr_set = 0;
  if (!lds_attr_set) {
    (void)hipFuncSetAttribute((const void*)recurrent_scan,
                              hipFuncAttributeMaxDynamicSharedMemorySize, DYN_BYTES);
    lds_attr_set = 1;
  }

  // reset sync counters (deterministic per call)
  zero_i32<<<1, 64, 0, stream>>>(sync, 8);

  // f32 -> f16 conversions (x + all weights)
  cvt_f32_to_f16<<<2048, 256, 0, stream>>>(x, seqA, MROWS * HDIM);
  for (int l = 0; l < 4; ++l) {
    cvt_f32_to_f16<<<1024, 256, 0, stream>>>(wih[l], wihh[l], GDIM * KDIM);
    cvt_f32_to_f16<<<1024, 256, 0, stream>>>(whh[l], whhh[l], NREC * KDIM);
  }

  float* out = (float*)d_out;
  float* hN  = out + (size_t)TSEQ * BATCH * HDIM;
  float* cN  = hN + (size_t)4 * BATCH * HDIM;

  _Float16* cur = seqA;
  _Float16* nxt = seqB;
  for (int l = 0; l < 4; ++l) {
    const int rev = l & 1;  // f0, b0, f1, b1

    dim3 g(MROWS / 64, GDIM / 64);
    gemm_xproj<<<g, 256, 0, stream>>>(cur, wihh[l], bih[l], xproj, MROWS, GDIM, KDIM);

    zero_f16<<<64, 256, 0, stream>>>(hq, BATCH * HDIM);

    recurrent_scan<<<32, 320, DYN_BYTES, stream>>>(
        xproj, masks, whhh[l], hq, nxt,
        (l == 3) ? out : nullptr,
        hN + (size_t)l * BATCH * HDIM,
        cN + (size_t)l * BATCH * HDIM,
        rev, sync, sync + 1);

    _Float16* tmp = cur; cur = nxt; nxt = tmp;
  }
}
